// VSN_22866405884382
// MI455X (gfx1250) — matrix operand order repaired and checked by reading, not compiled, not run on hardware
//
#include <hip/hip_runtime.h>
#include <hip/hip_bf16.h>

typedef __attribute__((ext_vector_type(16))) __bf16 v16bf;
typedef __attribute__((ext_vector_type(8)))  float  v8f;

#define VN 32
#define HN 256
#define BT 16384
#define EPSV 1e-5f

__device__ __forceinline__ float sigmoidf_(float x) { return 1.f / (1.f + __expf(-x)); }
__device__ __forceinline__ float eluf_(float x) { return (x > 0.f) ? x : (__expf(x) - 1.f); }

// ---------------------------------------------------------------------------
// Kernel 0: repack fc2_w [V,H,H] and glu_w [V,H,2H] (f32, row-major [k,n])
// into bf16 WMMA B-fragment layout: per (v, kt, nt) 32x16 tile, lane L holds
// 16 contiguous bf16: n = nt*16 + (L&15), k = kt*32 + (L>=16 ? 16 : 0) + e.
// ---------------------------------------------------------------------------
__global__ __launch_bounds__(256) void vsn_convert(const float* __restrict__ fc2_w,
                                                   const float* __restrict__ glu_w,
                                                   __bf16* __restrict__ B1,
                                                   __bf16* __restrict__ B2) {
  int idx = blockIdx.x * 256 + threadIdx.x;
  if (idx < 131072) {                       // fc2: 32v * 8kt * 16nt * 32lane
    int lane = idx & 31, nt = (idx >> 5) & 15, kt = (idx >> 9) & 7, v = idx >> 12;
    int n  = nt * 16 + (lane & 15);
    int kb = kt * 32 + ((lane >= 16) ? 16 : 0);
#pragma unroll
    for (int e = 0; e < 16; ++e)
      B1[(size_t)idx * 16 + e] = (__bf16)fc2_w[((size_t)v * HN + kb + e) * HN + n];
  } else {                                   // glu: 32v * 8kt * 32nt * 32lane
    int j = idx - 131072;
    if (j >= 262144) return;
    int lane = j & 31, nt = (j >> 5) & 31, kt = (j >> 10) & 7, v = j >> 13;
    int n  = nt * 16 + (lane & 15);
    int kb = kt * 32 + ((lane >= 16) ? 16 : 0);
#pragma unroll
    for (int e = 0; e < 16; ++e)
      B2[(size_t)j * 16 + e] = (__bf16)glu_w[((size_t)v * HN + kb + e) * (2 * HN) + n];
  }
}

// ---------------------------------------------------------------------------
// Kernel 1: weight GRN + softmax -> wsm [BT, V].  One wave32 per token
// (V == 32 == wave width, so LN/softmax over V are shfl reductions).
// ---------------------------------------------------------------------------
__global__ __launch_bounds__(256) void vsn_stage1(
    const float* __restrict__ x,
    const float* __restrict__ wf1_w, const float* __restrict__ wf1_b,
    const float* __restrict__ wf2_w, const float* __restrict__ wf2_b,
    const float* __restrict__ wglu_w, const float* __restrict__ wglu_b,
    const float* __restrict__ wln_g, const float* __restrict__ wln_b,
    float* __restrict__ wsm) {
  __shared__ float s_wf1[VN][HN];
  __shared__ float s_wf1b[HN];
  __shared__ float s_wf2[HN][VN];
  __shared__ float s_wf2b[VN];
  __shared__ float s_wglu[VN][2 * VN];
  __shared__ float s_wglub[2 * VN];
  __shared__ float s_lng[VN], s_lnb[VN];
  __shared__ float s_x[8][VN];
  __shared__ float s_h[8][HN];
  __shared__ float s_h2[8][VN];

  int tid = threadIdx.x;
  int lane = tid & 31, wave = tid >> 5;

  for (int i = tid; i < VN * HN; i += 256) s_wf1[i >> 8][i & 255] = wf1_w[i];
  for (int i = tid; i < HN * VN; i += 256) s_wf2[i >> 5][i & 31] = wf2_w[i];
  for (int i = tid; i < VN * 2 * VN; i += 256) s_wglu[i >> 6][i & 63] = wglu_w[i];
  if (tid < HN) s_wf1b[tid] = wf1_b[tid];
  if (tid < VN) { s_wf2b[tid] = wf2_b[tid]; s_lng[tid] = wln_g[tid]; s_lnb[tid] = wln_b[tid]; }
  if (tid < 2 * VN) s_wglub[tid] = wglu_b[tid];

  int t = blockIdx.x * 8 + wave;
  float xv = x[t * VN + lane];
  s_x[wave][lane] = xv;
  __syncthreads();

#pragma unroll
  for (int c = 0; c < 8; ++c) {            // h = elu(x @ wf1 + b), 8 k's/lane
    int k = lane + 32 * c;
    float s = s_wf1b[k];
#pragma unroll 8
    for (int v2 = 0; v2 < VN; ++v2) s += s_x[wave][v2] * s_wf1[v2][k];
    s_h[wave][k] = eluf_(s);
  }
  __syncthreads();

  float h2 = s_wf2b[lane];                 // h2 = h @ wf2 + b, lane = var
#pragma unroll 8
  for (int k = 0; k < HN; ++k) h2 += s_h[wave][k] * s_wf2[k][lane];
  s_h2[wave][lane] = h2;
  __syncthreads();

  float av = s_wglub[lane], gv = s_wglub[VN + lane];
#pragma unroll 8
  for (int v2 = 0; v2 < VN; ++v2) {
    float hh = s_h2[wave][v2];
    av += hh * s_wglu[v2][lane];
    gv += hh * s_wglu[v2][VN + lane];
  }
  float val = av * sigmoidf_(gv) + xv;     // GLU + identity skip

  float s1 = val;                          // layernorm over lanes (V)
#pragma unroll
  for (int off = 16; off >= 1; off >>= 1) s1 += __shfl_xor(s1, off);
  float mu = s1 * (1.f / 32.f);
  float d = val - mu;
  float s2 = d * d;
#pragma unroll
  for (int off = 16; off >= 1; off >>= 1) s2 += __shfl_xor(s2, off);
  float y = d * rsqrtf(s2 * (1.f / 32.f) + EPSV) * s_lng[lane] + s_lnb[lane];

  float mx = y;                            // softmax over lanes (V)
#pragma unroll
  for (int off = 16; off >= 1; off >>= 1) mx = fmaxf(mx, __shfl_xor(mx, off));
  float e = __expf(y - mx);
  float se = e;
#pragma unroll
  for (int off = 16; off >= 1; off >>= 1) se += __shfl_xor(se, off);
  wsm[t * VN + lane] = e / se;
}

// ---------------------------------------------------------------------------
// Kernel 2: fused per-variable GRNs + weighted sum.
// Block = 32-token tile (8 waves); loop over v; bf16 WMMA 16x16x32 GEMMs
// with A fragments staged in LDS, f32 accumulation, fused epilogues.
// ---------------------------------------------------------------------------
__global__ __launch_bounds__(256) void vsn_stage2(
    const float* __restrict__ x,
    const float* __restrict__ fc1_w, const float* __restrict__ fc1_b,
    const float* __restrict__ fc2_b, const float* __restrict__ glu_b,
    const float* __restrict__ ln_g, const float* __restrict__ ln_b,
    const float* __restrict__ skip_w, const float* __restrict__ skip_b,
    const float* __restrict__ wsm,
    const __bf16* __restrict__ B1, const __bf16* __restrict__ B2,
    float* __restrict__ out) {
  __shared__ float s_x[32][VN];            // token tile of x
  __shared__ float s_w[32][VN];            // softmax selection weights
  __shared__ float s_f1w[HN], s_f1b[HN];
  __shared__ v16bf s_A1[2][8][32];         // A frags for GEMM1 (elu(fc1))
  __shared__ v16bf s_A2[2][8][32];         // A frags for GEMM2 (h2)
  __shared__ float s_val[32][HN];          // pre-LN GLU output

  int tid = threadIdx.x;
  int lane = tid & 31, wave = tid >> 5;
  int tok0 = blockIdx.x * 32;

  for (int i = tid; i < 32 * VN; i += 256) {
    int m = i >> 5, vv = i & 31;
    s_x[m][vv] = x[(size_t)(tok0 + m) * VN + vv];
    s_w[m][vv] = wsm[(size_t)(tok0 + m) * VN + vv];
  }
  float accR[32];                          // wave owns tokens wave*4..+3
#pragma unroll
  for (int i = 0; i < 32; ++i) accR[i] = 0.f;
  __syncthreads();

  for (int v = 0; v < VN; ++v) {
    if (tid < HN) { s_f1w[tid] = fc1_w[v * HN + tid]; s_f1b[tid] = fc1_b[v * HN + tid]; }
    __syncthreads();

    // --- build A1 fragments: h1 = elu(x[:,v] * fc1_w[v] + fc1_b[v]), bf16 ---
    for (int idx = tid; idx < 512; idx += 256) {
      int fl = idx & 31, kt = (idx >> 5) & 7, mt = (idx >> 8) & 1;
      int m = mt * 16 + (fl & 15);
      int kbase = kt * 32 + ((fl >= 16) ? 16 : 0);
      float xm = s_x[m][v];
      v16bf frag;
#pragma unroll
      for (int e = 0; e < 16; ++e) {
        int k = kbase + e;
        frag[e] = (__bf16)eluf_(xm * s_f1w[k] + s_f1b[k]);
      }
      s_A1[mt][kt][fl] = frag;
    }
    __syncthreads();

    // --- GEMM1: h2 = h1 @ fc2_w[v] + fc2_b[v]  -> A2 fragments (bf16) ---
    for (int tIdx = wave; tIdx < 32; tIdx += 8) {
      int mt = tIdx >> 4, nt = tIdx & 15;
      v8f acc = {};
      const __bf16* bptr = B1 + (((size_t)(v * 8) * 16 + nt) * 32 + lane) * 16;
#pragma unroll
      for (int kt = 0; kt < 8; ++kt) {
        v16bf a = s_A1[mt][kt][lane];
        v16bf b = *(const v16bf*)(bptr + (size_t)kt * 8192);
        acc = __builtin_amdgcn_wmma_f32_16x16x32_bf16(false, a, false, b,
                                                      (short)0, acc, false, false);
      }
      int nc = nt * 16 + (lane & 15);
      float bias = fc2_b[v * HN + nc];
      // scatter D tile into A-fragment layout for GEMM2 (n becomes k)
      int kt2 = nt >> 1;
      int k32 = (nt & 1) * 16 + (lane & 15);
      int half = (k32 >> 4) & 1;
      int ep = k32 & 15;
      int mhi = (lane >= 16) ? 8 : 0;
#pragma unroll
      for (int j = 0; j < 8; ++j) {
        int lanep = (j + mhi) + half * 16;
        ((__bf16*)&s_A2[mt][kt2][lanep])[ep] = (__bf16)(acc[j] + bias);
      }
    }
    __syncthreads();

    // --- GEMM2: g = h2 @ glu_w[v]; GLU + residual, fused epilogue ---
    for (int tIdx = wave; tIdx < 32; tIdx += 8) {
      int mt = tIdx >> 4, nta = tIdx & 15;
      v8f acca = {}, accg = {};
      const __bf16* bptrA = B2 + (((size_t)(v * 8) * 32 + nta) * 32 + lane) * 16;
      const __bf16* bptrG = bptrA + 16 * 32 * 16;   // gate half: nt = nta + 16
#pragma unroll
      for (int kt = 0; kt < 8; ++kt) {
        v16bf a  = s_A2[mt][kt][lane];
        v16bf b0 = *(const v16bf*)(bptrA + (size_t)kt * 16384);
        v16bf b1 = *(const v16bf*)(bptrG + (size_t)kt * 16384);
        acca = __builtin_amdgcn_wmma_f32_16x16x32_bf16(false, a, false, b0,
                                                       (short)0, acca, false, false);
        accg = __builtin_amdgcn_wmma_f32_16x16x32_bf16(false, a, false, b1,
                                                       (short)0, accg, false, false);
      }
      int nc = nta * 16 + (lane & 15);
      float ba = glu_b[v * 2 * HN + nc];
      float bg = glu_b[v * 2 * HN + HN + nc];
      float sw = skip_w[v * HN + nc], sb = skip_b[v * HN + nc];
      int mhi = (lane >= 16) ? 8 : 0;
#pragma unroll
      for (int j = 0; j < 8; ++j) {
        int m = mt * 16 + j + mhi;
        float res = s_x[m][v] * sw + sb;
        s_val[m][nc] = (acca[j] + ba) * sigmoidf_(accg[j] + bg) + res;
      }
    }
    __syncthreads();

    // --- LayerNorm over H + softmax-weighted accumulate (regs) ---
#pragma unroll
    for (int mi = 0; mi < 4; ++mi) {
      int m = wave * 4 + mi;
      float s = 0.f, sq = 0.f;
#pragma unroll
      for (int c = 0; c < 8; ++c) { float t = s_val[m][lane + 32 * c]; s += t; sq += t * t; }
#pragma unroll
      for (int off = 16; off >= 1; off >>= 1) { s += __shfl_xor(s, off); sq += __shfl_xor(sq, off); }
      float mu = s * (1.f / 256.f);
      float rs = rsqrtf(sq * (1.f / 256.f) - mu * mu + EPSV);
      float wv = s_w[m][v];
#pragma unroll
      for (int c = 0; c < 8; ++c) {
        int n = lane + 32 * c;
        float y = (s_val[m][n] - mu) * rs * ln_g[v * HN + n] + ln_b[v * HN + n];
        accR[mi * 8 + c] += y * wv;
      }
    }
    __syncthreads();
  }

  // write out: wave owns its 4 tokens, lane strided over H
#pragma unroll
  for (int mi = 0; mi < 4; ++mi) {
    int m = wave * 4 + mi;
#pragma unroll
    for (int c = 0; c < 8; ++c)
      out[(size_t)(tok0 + m) * HN + lane + 32 * c] = accR[mi * 8 + c];
  }
}

extern "C" void kernel_launch(void* const* d_in, const int* in_sizes, int n_in,
                              void* d_out, int out_size, void* d_ws, size_t ws_size,
                              hipStream_t stream) {
  const float* x      = (const float*)d_in[0];
  const float* fc1_w  = (const float*)d_in[1];
  const float* fc1_b  = (const float*)d_in[2];
  const float* fc2_w  = (const float*)d_in[3];
  const float* fc2_b  = (const float*)d_in[4];
  const float* glu_w  = (const float*)d_in[5];
  const float* glu_b  = (const float*)d_in[6];
  const float* ln_g   = (const float*)d_in[7];
  const float* ln_b   = (const float*)d_in[8];
  const float* skip_w = (const float*)d_in[9];
  const float* skip_b = (const float*)d_in[10];
  const float* wf1_w  = (const float*)d_in[11];
  const float* wf1_b  = (const float*)d_in[12];
  const float* wf2_w  = (const float*)d_in[13];
  const float* wf2_b  = (const float*)d_in[14];
  const float* wglu_w = (const float*)d_in[15];
  const float* wglu_b = (const float*)d_in[16];
  const float* wln_g  = (const float*)d_in[17];
  const float* wln_b  = (const float*)d_in[18];
  float* out = (float*)d_out;

  // workspace layout: [0,2MB) softmax weights f32; [2MB,6MB) fc2 bf16 frags;
  // [6MB,14MB) glu bf16 frags.  Total 14 MB.
  float*  wsm = (float*)d_ws;
  __bf16* B1  = (__bf16*)((char*)d_ws + (size_t)(2u << 20));
  __bf16* B2  = (__bf16*)((char*)d_ws + (size_t)(6u << 20));

  vsn_convert<<<1536, 256, 0, stream>>>(fc2_w, glu_w, B1, B2);
  vsn_stage1<<<BT / 8, 256, 0, stream>>>(x, wf1_w, wf1_b, wf2_w, wf2_b,
                                         wglu_w, wglu_b, wln_g, wln_b, wsm);
  vsn_stage2<<<BT / 32, 256, 0, stream>>>(x, fc1_w, fc1_b, fc2_b, glu_b,
                                          ln_g, ln_b, skip_w, skip_b,
                                          wsm, B1, B2, out);
}
